// SimpleNeuroSAT_77884936945774
// MI455X (gfx1250) — matrix operand order repaired and checked by reading, not compiled, not run on hardware
//
#include <hip/hip_runtime.h>
#include <hip/hip_bf16.h>

#define FM        80
#define N_VARS    100000
#define N_LITS    200000
#define N_CLAUSES 420000
#define N_CELLS   (N_CLAUSES * 3)
#define N_GRAPHS  100
#define CPG       (N_CLAUSES / N_GRAPHS)   /* 4200 */
#define NROUNDS   32

typedef __attribute__((ext_vector_type(16))) __bf16 bf16x16;
typedef __attribute__((ext_vector_type(8)))  float  floatx8;

// ---------------------------------------------------------------------------
// WMMA fragment helpers (CDNA5 v_wmma_f32_16x16x32_bf16, wave32)
// A 16x32 bf16: lane m=l&15; lanes0-15: K=0..7,16..23 ; lanes16-31: K=8..15,24..31
// B 32x16 bf16: lane n=l&15; lanes0-15: K=0..7,16..23 ; lanes16-31: K=8..15,24..31 (same as A)
// C/D f32 16x16: lane n=l&15; VGPR r -> M = r + 8*(l>>4)
// ---------------------------------------------------------------------------
__device__ __forceinline__ bf16x16 frag_a(const __bf16* Xw, int ldx, int k0, int lane) {
  const int m    = lane & 15;
  const int koff = (lane >> 4) << 3;
  const __bf16* p = Xw + m * ldx + k0 + koff;
  bf16x16 a;
#pragma unroll
  for (int i = 0; i < 8; ++i) { a[i] = p[i]; a[i + 8] = p[i + 16]; }
  return a;
}

__device__ __forceinline__ bf16x16 frag_b(const __bf16* Wt, int K, int n0, int k0, int lane) {
  const int n  = lane & 15;
  const int kb = (lane >> 4) << 3;
  const __bf16* q = Wt + (size_t)(n0 + n) * K + k0 + kb;
  bf16x16 b;
#pragma unroll
  for (int i = 0; i < 8; ++i) { b[i] = q[i]; b[i + 8] = q[i + 16]; }
  return b;
}

__device__ __forceinline__ floatx8 wmma_bf16(bf16x16 a, bf16x16 b, floatx8 c) {
  return __builtin_amdgcn_wmma_f32_16x16x32_bf16(false, a, false, b, (short)0, c, false, false);
}

// 16-row GEMM tile: LDS bf16 X -> LDS bf16 H, bias + relu6
__device__ __forceinline__ void layer_lds(const __bf16* Xw, int ldx, int K,
                                          const __bf16* __restrict__ Wt,
                                          const float* __restrict__ bias,
                                          int N, __bf16* Hw, int ldh, int lane) {
  for (int n0 = 0; n0 < N; n0 += 16) {
    floatx8 acc;
#pragma unroll
    for (int i = 0; i < 8; ++i) acc[i] = 0.0f;
#pragma unroll 2
    for (int k0 = 0; k0 < K; k0 += 32)
      acc = wmma_bf16(frag_a(Xw, ldx, k0, lane), frag_b(Wt, K, n0, k0, lane), acc);
    const int n  = lane & 15;
    const int mb = (lane >> 4) << 3;
    const float bv = bias[n0 + n];
#pragma unroll
    for (int r = 0; r < 8; ++r) {
      float v = acc[r] + bv;
      v = fminf(fmaxf(v, 0.0f), 6.0f);          // relu6
      Hw[(mb + r) * ldh + n0 + n] = (__bf16)v;
    }
  }
}

// 16-row GEMM tile: LDS bf16 -> global f32 (no activation), row-guarded.
// Also accumulates per-column sum / sum-of-squares into block-shared arrays
// (fused batch-norm statistics: saves a full re-read of the output array).
__device__ __forceinline__ void layer_glb_stats(const __bf16* Xw, int ldx, int K,
                                                const __bf16* __restrict__ Wt,
                                                const float* __restrict__ bias,
                                                int N, float* __restrict__ out,
                                                int grow0, int nrows,
                                                float* colsum, float* colsq, int lane) {
  for (int n0 = 0; n0 < N; n0 += 16) {
    floatx8 acc;
#pragma unroll
    for (int i = 0; i < 8; ++i) acc[i] = 0.0f;
#pragma unroll 2
    for (int k0 = 0; k0 < K; k0 += 32)
      acc = wmma_bf16(frag_a(Xw, ldx, k0, lane), frag_b(Wt, K, n0, k0, lane), acc);
    const int n  = lane & 15;
    const int mb = (lane >> 4) << 3;
    const float bv = bias[n0 + n];
    float s = 0.f, q = 0.f;
#pragma unroll
    for (int r = 0; r < 8; ++r) {
      int row = grow0 + mb + r;
      float v = acc[r] + bv;
      if (row < nrows) {
        out[(size_t)row * N + n0 + n] = v;
        s += v; q += v * v;
      }
    }
    atomicAdd(&colsum[n0 + n], s);   // LDS atomic (ds_add_f32)
    atomicAdd(&colsq[n0 + n], q);
  }
}

// ---------------------------------------------------------------------------
// Setup kernels
// ---------------------------------------------------------------------------
__global__ void k_prep(const int* __restrict__ lit_var, const int* __restrict__ lit_neg,
                       int* __restrict__ lit_row, float* __restrict__ sgn) {
  int i = blockIdx.x * blockDim.x + threadIdx.x;
  if (i < N_CELLS) {
    int neg = lit_neg[i];
    lit_row[i] = lit_var[i] + neg * N_VARS;
    sgn[i] = 1.0f - 2.0f * (float)neg;
  }
}

__global__ void k_fill(float* __restrict__ p, size_t n, const float* __restrict__ scale) {
  size_t i = blockIdx.x * (size_t)blockDim.x + threadIdx.x;
  if (i < n) p[i] = scale[0];
}

// f32 W[K][N] -> bf16 Wt[N][Kpad] (K-major, zero-padded)
__global__ void k_convwt(const float* __restrict__ W, __bf16* __restrict__ Wt,
                         int K, int Kpad, int N) {
  int i = blockIdx.x * blockDim.x + threadIdx.x;
  if (i < N * Kpad) {
    int n = i / Kpad, k = i % Kpad;
    Wt[i] = (k < K) ? (__bf16)W[(size_t)k * N + n] : (__bf16)0.0f;
  }
}

// ---------------------------------------------------------------------------
// Message passing: gather / scatter
// ---------------------------------------------------------------------------
__global__ void k_lc(const float* __restrict__ L, const int* __restrict__ lit_row,
                     const float* __restrict__ scale_p, float* __restrict__ LC) {
  int i = blockIdx.x * blockDim.x + threadIdx.x;
  if (i >= N_CLAUSES * (FM / 4)) return;
  int c = i / (FM / 4), f = (i % (FM / 4)) * 4;
  int r0 = lit_row[3 * c], r1 = lit_row[3 * c + 1], r2 = lit_row[3 * c + 2];
  float4 a = *(const float4*)(L + (size_t)r0 * FM + f);
  float4 b = *(const float4*)(L + (size_t)r1 * FM + f);
  float4 d = *(const float4*)(L + (size_t)r2 * FM + f);
  float s = scale_p[0];
  float4 o;
  o.x = (a.x + b.x + d.x) * s; o.y = (a.y + b.y + d.y) * s;
  o.z = (a.z + b.z + d.z) * s; o.w = (a.w + b.w + d.w) * s;
  *(float4*)(LC + (size_t)c * FM + f) = o;
}

__global__ void k_cl(const float* __restrict__ C, const int* __restrict__ lit_row,
                     const float* __restrict__ scale_p, float* __restrict__ CL) {
  int i = blockIdx.x * blockDim.x + threadIdx.x;
  if (i >= N_CELLS * (FM / 4)) return;
  int cell = i / (FM / 4), f = (i % (FM / 4)) * 4;
  int clause = cell / 3;
  int row = lit_row[cell];
  float s = scale_p[0];
  float4 v = *(const float4*)(C + (size_t)clause * FM + f);
  float* d = CL + (size_t)row * FM + f;
  unsafeAtomicAdd(d + 0, v.x * s);
  unsafeAtomicAdd(d + 1, v.y * s);
  unsafeAtomicAdd(d + 2, v.z * s);
  unsafeAtomicAdd(d + 3, v.w * s);
}

// ---------------------------------------------------------------------------
// WMMA MLP kernels (with fused batch-norm statistics)
// ---------------------------------------------------------------------------
__global__ __launch_bounds__(128) void k_mlp_c(const float* __restrict__ C,
                                               const float* __restrict__ LC,
                                               const __bf16* __restrict__ W0t, const float* __restrict__ b0,
                                               const __bf16* __restrict__ W1t, const float* __restrict__ b1,
                                               float* __restrict__ Craw, float* __restrict__ stats) {
  __shared__ __bf16 X[64 * 168];
  __shared__ __bf16 H[64 * 168];
  __shared__ float CS[FM], CQ[FM];
  const int tid = threadIdx.x;
  const int row0 = blockIdx.x * 64;
  if (tid < FM) { CS[tid] = 0.f; CQ[tid] = 0.f; }
  for (int i = tid; i < 64 * FM; i += 128) {
    int r = i / FM, c = i % FM;
    int gr = row0 + r;
    float v0 = 0.f, v1 = 0.f;
    if (gr < N_CLAUSES) { v0 = C[(size_t)gr * FM + c]; v1 = LC[(size_t)gr * FM + c]; }
    X[r * 168 + c]      = (__bf16)v0;
    X[r * 168 + FM + c] = (__bf16)v1;
  }
  __syncthreads();
  const int wave = tid >> 5, lane = tid & 31;
  const __bf16* Xw = X + wave * 16 * 168;
  __bf16* Hw = H + wave * 16 * 168;
  layer_lds(Xw, 168, 160, W0t, b0, 160, Hw, 168, lane);
  __syncthreads();
  layer_glb_stats(Hw, 168, 160, W1t, b1, 80, Craw, row0 + wave * 16, N_CLAUSES,
                  CS, CQ, lane);
  __syncthreads();
  if (tid < FM) {
    unsafeAtomicAdd(&stats[tid], CS[tid]);
    unsafeAtomicAdd(&stats[FM + tid], CQ[tid]);
  }
}

__global__ __launch_bounds__(64) void k_mlp_l(const float* __restrict__ L,
                                              const float* __restrict__ CL,
                                              const __bf16* __restrict__ W0t, const float* __restrict__ b0,
                                              const __bf16* __restrict__ W1t, const float* __restrict__ b1,
                                              float* __restrict__ Lraw, float* __restrict__ stats) {
  __shared__ __bf16 X[32 * 264];   // 240 used + pad; K padded to 256
  __shared__ __bf16 H[32 * 264];
  __shared__ float CS[FM], CQ[FM];
  const int tid = threadIdx.x;
  const int row0 = blockIdx.x * 32;
  for (int i = tid; i < FM; i += 64) { CS[i] = 0.f; CQ[i] = 0.f; }
  for (int i = tid; i < 32 * FM; i += 64) {
    int r = i / FM, c = i % FM;
    int gr = row0 + r;
    float v0 = 0.f, v1 = 0.f, v2 = 0.f;
    if (gr < N_LITS) {
      v0 = L[(size_t)gr * FM + c];
      v1 = CL[(size_t)gr * FM + c];
      int fr = (gr < N_VARS) ? gr + N_VARS : gr - N_VARS;  // flip(L)
      v2 = L[(size_t)fr * FM + c];
    }
    X[r * 264 + c]          = (__bf16)v0;
    X[r * 264 + FM + c]     = (__bf16)v1;
    X[r * 264 + 2 * FM + c] = (__bf16)v2;
  }
  for (int i = tid; i < 32 * 16; i += 64) {   // zero K-pad cols 240..255
    int r = i / 16, c = 240 + (i % 16);
    X[r * 264 + c] = (__bf16)0.0f;
    H[r * 264 + c] = (__bf16)0.0f;
  }
  __syncthreads();
  const int wave = tid >> 5, lane = tid & 31;
  const __bf16* Xw = X + wave * 16 * 264;
  __bf16* Hw = H + wave * 16 * 264;
  layer_lds(Xw, 264, 256, W0t, b0, 240, Hw, 264, lane);
  __syncthreads();
  layer_glb_stats(Hw, 264, 256, W1t, b1, 80, Lraw, row0 + wave * 16, N_LITS,
                  CS, CQ, lane);
  __syncthreads();
  if (tid < FM) {
    unsafeAtomicAdd(&stats[tid], CS[tid]);
    unsafeAtomicAdd(&stats[FM + tid], CQ[tid]);
  }
}

__global__ __launch_bounds__(128) void k_vscore(const float* __restrict__ L,
      const __bf16* __restrict__ V0t, const float* __restrict__ vb0,
      const __bf16* __restrict__ V1t, const float* __restrict__ vb1,
      const __bf16* __restrict__ V2t, const float* __restrict__ vb2,
      const __bf16* __restrict__ V3t, const float* __restrict__ vb3,
      float* __restrict__ logits) {
  __shared__ __bf16 X[64 * 168];
  __shared__ __bf16 H[64 * 168];
  const int tid = threadIdx.x;
  const int row0 = blockIdx.x * 64;
  for (int i = tid; i < 64 * FM; i += 128) {
    int r = i / FM, c = i % FM;
    int gr = row0 + r;
    float v0 = 0.f, v1 = 0.f;
    if (gr < N_VARS) { v0 = L[(size_t)gr * FM + c]; v1 = L[(size_t)(gr + N_VARS) * FM + c]; }
    X[r * 168 + c]      = (__bf16)v0;
    X[r * 168 + FM + c] = (__bf16)v1;
  }
  __syncthreads();
  const int wave = tid >> 5, lane = tid & 31;
  __bf16* Xw = X + wave * 16 * 168;
  __bf16* Hw = H + wave * 16 * 168;
  layer_lds(Xw, 168, 160, V0t, vb0, 160, Hw, 168, lane);
  __syncthreads();
  layer_lds(Hw, 168, 160, V1t, vb1, 160, Xw, 168, lane);  // ping-pong
  __syncthreads();
  layer_lds(Xw, 168, 160, V2t, vb2, 160, Hw, 168, lane);
  __syncthreads();
  // final 160 -> 1 dot: 2 lanes per row, halves combined via wave shuffle
  {
    int r    = tid >> 1;          // 0..63
    int half = tid & 1;
    const __bf16* h = H + r * 168 + half * 80;
    const __bf16* w = V3t + half * 80;
    float s = 0.f;
#pragma unroll 8
    for (int k = 0; k < 80; ++k) s += (float)h[k] * (float)w[k];
    s += __shfl_xor(s, 1, 32);    // partner lane holds the other half
    int gr = row0 + r;
    if (half == 0 && gr < N_VARS) logits[gr] = s + vb3[0];
  }
}

// ---------------------------------------------------------------------------
// Column-wise batch norm apply (axis 0), eps = 1e-3
// ---------------------------------------------------------------------------
__global__ void k_colnorm(const float* __restrict__ Xr, const float* __restrict__ stats,
                          int nrows, float* __restrict__ Y) {
  size_t i = blockIdx.x * (size_t)blockDim.x + threadIdx.x;
  size_t tot = (size_t)nrows * FM;
  if (i >= tot) return;
  int c = (int)(i % FM);
  float inv_n = 1.0f / (float)nrows;
  float mean = stats[c] * inv_n;
  float var  = stats[FM + c] * inv_n - mean * mean;
  Y[i] = (Xr[i] - mean) * rsqrtf(var + 1e-3f);
}

// ---------------------------------------------------------------------------
// Loss
// ---------------------------------------------------------------------------
__global__ __launch_bounds__(256) void k_loss(const float* __restrict__ logits,
                                              const int* __restrict__ lit_var,
                                              const float* __restrict__ sgn,
                                              float* __restrict__ per_graph) {
  __shared__ float acc2[2];
  const int tid = threadIdx.x;
  if (tid < 2) acc2[tid] = 0.f;
  __syncthreads();
  const int c  = blockIdx.x * 256 + tid;
  const int g0 = (blockIdx.x * 256) / CPG;   // block spans at most 2 graphs
  if (c < N_CLAUSES) {
    float ssum = 0.f;
#pragma unroll
    for (int j = 0; j < 3; ++j) {
      int cell = 3 * c + j;
      float x = logits[lit_var[cell]] * sgn[cell];
      float sp = (x > 0.f) ? (x + log1pf(expf(-x))) : log1pf(expf(x));  // softplus
      ssum += sp;
    }
    float vc = expf(-ssum);
    float pc = vc * (-logf(1.0f - vc + 1e-8f));
    atomicAdd(&acc2[(c / CPG) - g0], pc);    // LDS atomic
  }
  __syncthreads();
  if (tid < 2) {
    int g = g0 + tid;
    if (g < N_GRAPHS && acc2[tid] != 0.f) unsafeAtomicAdd(&per_graph[g], acc2[tid]);
  }
}

__global__ void k_lred(const float* __restrict__ per_graph, float* __restrict__ loss_acc) {
  __shared__ float sm[128];
  int t = threadIdx.x;
  sm[t] = (t < N_GRAPHS) ? sqrtf(per_graph[t] + 1e-6f) : 0.f;
  __syncthreads();
  for (int s = 64; s > 0; s >>= 1) { if (t < s) sm[t] += sm[t + s]; __syncthreads(); }
  if (t == 0) loss_acc[0] += sm[0];
}

__global__ void k_final(const float* __restrict__ logits, const float* __restrict__ loss_acc,
                        float* __restrict__ out) {
  int i = blockIdx.x * blockDim.x + threadIdx.x;
  if (i < N_VARS) out[i] = logits[i];
  else if (i == N_VARS) out[i] = loss_acc[0] / (float)NROUNDS;
}

// ---------------------------------------------------------------------------
extern "C" void kernel_launch(void* const* d_in, const int* in_sizes, int n_in,
                              void* d_out, int out_size, void* d_ws, size_t ws_size,
                              hipStream_t stream) {
  (void)in_sizes; (void)n_in; (void)out_size; (void)ws_size;
  const int*   lit_var = (const int*)d_in[0];
  const int*   lit_neg = (const int*)d_in[1];
  const float* Lscale  = (const float*)d_in[7];
  const float* Cscale  = (const float*)d_in[8];
  const float* LCs     = (const float*)d_in[9];
  const float* CLs     = (const float*)d_in[10];
  const float* LuW0 = (const float*)d_in[11]; const float* Lub0 = (const float*)d_in[12];
  const float* LuW1 = (const float*)d_in[13]; const float* Lub1 = (const float*)d_in[14];
  const float* CuW0 = (const float*)d_in[15]; const float* Cub0 = (const float*)d_in[16];
  const float* CuW1 = (const float*)d_in[17]; const float* Cub1 = (const float*)d_in[18];
  const float* VsW0 = (const float*)d_in[19]; const float* Vsb0 = (const float*)d_in[20];
  const float* VsW1 = (const float*)d_in[21]; const float* Vsb1 = (const float*)d_in[22];
  const float* VsW2 = (const float*)d_in[23]; const float* Vsb2 = (const float*)d_in[24];
  const float* VsW3 = (const float*)d_in[25]; const float* Vsb3 = (const float*)d_in[26];

  char* p = (char*)d_ws;
  auto alloc = [&](size_t bytes) { char* r = p; p += (bytes + 255) & ~(size_t)255; return r; };
  float*  L      = (float*)alloc((size_t)N_LITS * FM * 4);
  float*  C      = (float*)alloc((size_t)N_CLAUSES * FM * 4);
  float*  RAW    = (float*)alloc((size_t)N_CLAUSES * FM * 4);  // shared C_raw / L_raw
  float*  LC     = (float*)alloc((size_t)N_CLAUSES * FM * 4);
  float*  CL     = (float*)alloc((size_t)N_LITS * FM * 4);
  float*  logits = (float*)alloc((size_t)N_VARS * 4);
  float*  stats  = (float*)alloc(2 * FM * 4);
  float*  per_g  = (float*)alloc(128 * 4);
  float*  lacc   = (float*)alloc(256);
  int*    lrow   = (int*)alloc((size_t)N_CELLS * 4);
  float*  sgn    = (float*)alloc((size_t)N_CELLS * 4);
  __bf16* W0c = (__bf16*)alloc(160 * 160 * 2);
  __bf16* W1c = (__bf16*)alloc(80 * 160 * 2);
  __bf16* W0l = (__bf16*)alloc(240 * 256 * 2);
  __bf16* W1l = (__bf16*)alloc(80 * 256 * 2);
  __bf16* V0  = (__bf16*)alloc(160 * 160 * 2);
  __bf16* V1  = (__bf16*)alloc(160 * 160 * 2);
  __bf16* V2  = (__bf16*)alloc(160 * 160 * 2);
  __bf16* V3  = (__bf16*)alloc(160 * 2);

  // one-time setup (re-run every call: deterministic)
  hipMemsetAsync(lacc, 0, 4, stream);
  k_prep<<<(N_CELLS + 255) / 256, 256, 0, stream>>>(lit_var, lit_neg, lrow, sgn);
  k_convwt<<<(160 * 160 + 255) / 256, 256, 0, stream>>>(CuW0, W0c, 160, 160, 160);
  k_convwt<<<(80 * 160 + 255) / 256, 256, 0, stream>>>(CuW1, W1c, 160, 160, 80);
  k_convwt<<<(240 * 256 + 255) / 256, 256, 0, stream>>>(LuW0, W0l, 240, 256, 240);
  k_convwt<<<(80 * 256 + 255) / 256, 256, 0, stream>>>(LuW1, W1l, 240, 256, 80);
  k_convwt<<<(160 * 160 + 255) / 256, 256, 0, stream>>>(VsW0, V0, 160, 160, 160);
  k_convwt<<<(160 * 160 + 255) / 256, 256, 0, stream>>>(VsW1, V1, 160, 160, 160);
  k_convwt<<<(160 * 160 + 255) / 256, 256, 0, stream>>>(VsW2, V2, 160, 160, 160);
  k_convwt<<<(160 + 255) / 256, 256, 0, stream>>>(VsW3, V3, 160, 160, 1);
  k_fill<<<((size_t)N_LITS * FM + 255) / 256, 256, 0, stream>>>(L, (size_t)N_LITS * FM, Lscale);
  k_fill<<<((size_t)N_CLAUSES * FM + 255) / 256, 256, 0, stream>>>(C, (size_t)N_CLAUSES * FM, Cscale);

  for (int round = 0; round < NROUNDS; ++round) {
    // clause messages + C update (norm stats fused into MLP epilogue)
    k_lc<<<(N_CLAUSES * (FM / 4) + 255) / 256, 256, 0, stream>>>(L, lrow, LCs, LC);
    hipMemsetAsync(stats, 0, 2 * FM * 4, stream);
    k_mlp_c<<<(N_CLAUSES + 63) / 64, 128, 0, stream>>>(C, LC, W0c, Cub0, W1c, Cub1, RAW, stats);
    k_colnorm<<<((size_t)N_CLAUSES * FM + 255) / 256, 256, 0, stream>>>(RAW, stats, N_CLAUSES, C);
    // literal messages + L update
    hipMemsetAsync(CL, 0, (size_t)N_LITS * FM * 4, stream);
    k_cl<<<(N_CELLS * (FM / 4) + 255) / 256, 256, 0, stream>>>(C, lrow, CLs, CL);
    hipMemsetAsync(stats, 0, 2 * FM * 4, stream);
    k_mlp_l<<<(N_LITS + 31) / 32, 64, 0, stream>>>(L, CL, W0l, Lub0, W1l, Lub1, RAW, stats);
    k_colnorm<<<((size_t)N_LITS * FM + 255) / 256, 256, 0, stream>>>(RAW, stats, N_LITS, L);
    // scoring + loss
    k_vscore<<<(N_VARS + 63) / 64, 128, 0, stream>>>(L, V0, Vsb0, V1, Vsb1, V2, Vsb2, V3, Vsb3, logits);
    hipMemsetAsync(per_g, 0, 128 * 4, stream);
    k_loss<<<(N_CLAUSES + 255) / 256, 256, 0, stream>>>(logits, lit_var, sgn, per_g);
    k_lred<<<1, 128, 0, stream>>>(per_g, lacc);
  }
  k_final<<<(N_VARS + 256) / 256, 256, 0, stream>>>(logits, lacc, (float*)d_out);
}
